// BiRNN_60266981098079
// MI455X (gfx1250) — matrix operand order repaired and checked by reading, not compiled, not run on hardware
//
#include <hip/hip_runtime.h>

typedef __attribute__((ext_vector_type(16))) _Float16 v16h;
typedef __attribute__((ext_vector_type(8)))  _Float16 v8h;
typedef __attribute__((ext_vector_type(8)))  float    v8f;

#define T_LEN 2048
#define B_SZ  256
#define H_SZ  64
#define WPS   96   // W' LDS row stride (f16 elems): K = 0..95 (64 hh + x + bias + pad)
#define FCS   64   // W_fc LDS row stride
#define HBS   72   // H tile LDS row stride (16B-aligned b128, bank-split halves)

static __device__ __forceinline__ float tanh_fast(float x) {
#if __has_builtin(__builtin_amdgcn_tanhf)
  return __builtin_amdgcn_tanhf(x);            // native v_tanh_f32 on gfx1250
#else
  float ax = __builtin_fabsf(x);
#if __has_builtin(__builtin_amdgcn_exp2f)
  float e  = __builtin_amdgcn_exp2f(ax * -2.8853900432586670f);  // exp(-2ax)
#else
  float e  = __expf(-2.0f * ax);
#endif
  // (1-e)/(1+e) = 2/(1+e) - 1
  float t  = __builtin_fmaf(2.0f, __builtin_amdgcn_rcpf(1.0f + e), -1.0f);
  return __builtin_copysignf(t, x);
#endif
}

static __device__ __forceinline__ float sigmoid_fast(float z) {
#if __has_builtin(__builtin_amdgcn_tanhf)
  return __builtin_fmaf(0.5f, __builtin_amdgcn_tanhf(0.5f * z), 0.5f);
#else
  float e = __expf(-z);
  return __builtin_amdgcn_rcpf(1.0f + e);
#endif
}

// B fragment (K x 16 tile, 16-bit): lane l holds N = n*16 + (l&15);
// lanes 0-15 take K = 32c..32c+15, lanes 16-31 take K = 32c+16..32c+31,
// packed 2 per VGPR. Source LDS is [N][K] row-major -> two b128 loads.
static __device__ __forceinline__ v16h load_frag_B(const _Float16* base, int stride,
                                                   int n, int c, int lane) {
  int col = n * 16 + (lane & 15);
  int k0  = 32 * c + ((lane >> 4) << 4);
  const v8h* p = (const v8h*)(base + col * stride + k0);
  v8h lo = p[0];
  v8h hi = p[1];
  return __builtin_shufflevector(lo, hi, 0,1,2,3,4,5,6,7,8,9,10,11,12,13,14,15);
}

// A fragment (16 x 32, 16-bit): lane l holds M = l&15;
// VGPR0-3: K = 32c + 16*(l>=16) + 0..7 ; VGPR4-7: K = 32c + 16*(l>=16) + 8..15
static __device__ __forceinline__ v16h load_frag_A(const _Float16* hbuf, int c, int lane) {
  int m = lane & 15;
  int g = (lane >> 4) << 4;
  const v8h* p0 = (const v8h*)(hbuf + m * HBS + 32 * c + g);
  const v8h* p1 = (const v8h*)(hbuf + m * HBS + 32 * c + 8 + g);
  return __builtin_shufflevector(p0[0], p1[0], 0,1,2,3,4,5,6,7,8,9,10,11,12,13,14,15);
}

__global__ __launch_bounds__(32)
void birnn_scan_kernel(const float* __restrict__ x,
                       const float* __restrict__ w_ih_f, const float* __restrict__ w_hh_f,
                       const float* __restrict__ b_ih_f, const float* __restrict__ b_hh_f,
                       const float* __restrict__ w_ih_b, const float* __restrict__ w_hh_b,
                       const float* __restrict__ b_ih_b, const float* __restrict__ b_hh_b,
                       const float* __restrict__ w_fc,
                       float* __restrict__ s_ws,     // [2][T][B] scalars y_dir . w_fc
                       float* __restrict__ hn_out)   // [2][B][H] final hidden states
{
  __shared__ _Float16 Wp [H_SZ * WPS];   // augmented W' : [n][k], k<64 = w_hh[n][k], 64=w_ih, 65=bias
  __shared__ _Float16 Wfc[16   * FCS];   // [n][k], row 0 = w_fc, rest 0
  __shared__ _Float16 Hb [16   * HBS];   // current H tile [m][j] in f16

  const int lane = threadIdx.x;
  const int dir  = blockIdx.x >> 4;     // 0 = forward, 1 = backward
  const int tile = blockIdx.x & 15;
  const int R    = tile * 16;           // batch-row base

  const float* w_ih = dir ? w_ih_b : w_ih_f;
  const float* w_hh = dir ? w_hh_b : w_hh_f;
  const float* b_i  = dir ? b_ih_b : b_ih_f;
  const float* b_h  = dir ? b_hh_b : b_hh_f;

  // ---- one-time: build W' and Wfc in LDS ----
  for (int idx = lane; idx < H_SZ * WPS; idx += 32) {
    int n = idx / WPS, k = idx - n * WPS;
    float v = 0.0f;
    if (k < 64)       v = w_hh[n * 64 + k];   // B[k][n] = W_hh[n][k]
    else if (k == 64) v = w_ih[n];
    else if (k == 65) v = b_i[n] + b_h[n];
    Wp[idx] = (_Float16)v;
  }
  for (int idx = lane; idx < 16 * FCS; idx += 32) {
    int n = idx / FCS, k = idx - n * FCS;
    Wfc[idx] = (n == 0) ? (_Float16)w_fc[k] : (_Float16)0.0f;
  }
  __syncthreads();
  asm volatile("s_wait_dscnt 0x0" ::: "memory");

  // ---- load constant B fragments (held in VGPRs for the whole scan) ----
  v16h Bw[3][4];
#pragma unroll
  for (int c = 0; c < 3; ++c)
#pragma unroll
    for (int n = 0; n < 4; ++n)
      Bw[c][n] = load_frag_B(Wp, WPS, n, c, lane);
  v16h Bf[2];
#pragma unroll
  for (int c = 0; c < 2; ++c)
    Bf[c] = load_frag_B(Wfc, FCS, 0, c, lane);

  // ---- sequential scan ----
  v16h A0 = {}, A1 = {};                   // h0 = 0
  v8f  hlast[4] = {};
  const float* xrow  = x + (R + (lane & 15)) * T_LEN;
  float*       s_dir = s_ws + dir * (T_LEN * B_SZ);
  const int    ghalf = (lane >> 4) << 3;   // 0 or 8 (M offset of this half-wave)

  // software-pipelined x load: xv_next always holds x for the upcoming step
  float xv_next = xrow[dir ? (T_LEN - 1) : 0];

#pragma unroll 1
  for (int step = 0; step < T_LEN; ++step) {
    const int t  = dir ? (T_LEN - 1 - step) : step;
    // issue next step's x load now; its wait lands one full step later
    const int tn = dir ? (t > 0 ? t - 1 : 0) : (t < T_LEN - 1 ? t + 1 : t);
    float xv = xv_next;
    xv_next = xrow[tn];
    const int tp = dir ? (t >= 64 ? t - 64 : 0) : (t + 64 < T_LEN ? t + 64 : T_LEN - 1);
    __builtin_prefetch(xrow + tp, 0, 0);   // global_prefetch_b8 on the x stream

    // augmented A chunk 2: K=64 -> x_t, K=65 -> 1.0 (lanes 0-15 only)
    v16h A2 = {};
    A2[0] = (lane < 16) ? (_Float16)xv   : (_Float16)0.0f;
    A2[1] = (lane < 16) ? (_Float16)1.0f : (_Float16)0.0f;

    // H_new = tanh( H @ W_hh^T + x*w_ih + bias ) : 12 WMMAs, f32 accumulate
    v8f acc[4];
#pragma unroll
    for (int n = 0; n < 4; ++n) {
      v8f c = {};
      c = __builtin_amdgcn_wmma_f32_16x16x32_f16(false, A0, false, Bw[0][n], (short)0, c, false, false);
      c = __builtin_amdgcn_wmma_f32_16x16x32_f16(false, A1, false, Bw[1][n], (short)0, c, false, false);
      c = __builtin_amdgcn_wmma_f32_16x16x32_f16(false, A2, false, Bw[2][n], (short)0, c, false, false);
      acc[n] = c;
    }

    // tanh + write H_new (f16) into LDS in [m][j] order
#pragma unroll
    for (int n = 0; n < 4; ++n) {
#pragma unroll
      for (int v = 0; v < 8; ++v) {
        float h = tanh_fast(acc[n][v]);
        acc[n][v] = h;
        Hb[(v + ghalf) * HBS + 16 * n + (lane & 15)] = (_Float16)h;
      }
      hlast[n] = acc[n];
    }
    asm volatile("s_wait_dscnt 0x0" ::: "memory");

    // reload H_new as A fragments (used by fc now, and by the recurrence next step)
    A0 = load_frag_A(Hb, 0, lane);
    A1 = load_frag_A(Hb, 1, lane);

    // s = H_new . w_fc via 2 WMMAs (result lives in N==0 column -> lanes 0 and 16)
    v8f f = {};
    f = __builtin_amdgcn_wmma_f32_16x16x32_f16(false, A0, false, Bf[0], (short)0, f, false, false);
    f = __builtin_amdgcn_wmma_f32_16x16x32_f16(false, A1, false, Bf[1], (short)0, f, false, false);

    if ((lane & 15) == 0) {
      float* sp = s_dir + t * B_SZ + R + ghalf;
      float4 lo = make_float4(f[0], f[1], f[2], f[3]);
      float4 hi = make_float4(f[4], f[5], f[6], f[7]);
      *reinterpret_cast<float4*>(sp)     = lo;
      *reinterpret_cast<float4*>(sp + 4) = hi;
    }
  }

  // ---- final hidden state h_n[dir][b][j] ----
#pragma unroll
  for (int n = 0; n < 4; ++n)
#pragma unroll
    for (int v = 0; v < 8; ++v) {
      int m = v + ghalf;
      hn_out[dir * (B_SZ * H_SZ) + (R + m) * H_SZ + 16 * n + (lane & 15)] = hlast[n][v];
    }
}

__global__ __launch_bounds__(256)
void birnn_combine_kernel(const float* __restrict__ s_ws,
                          const float* __restrict__ b_fc,
                          float* __restrict__ y) {
  int i = blockIdx.x * blockDim.x + threadIdx.x;
  if (i >= B_SZ * T_LEN) return;
  int b = i / T_LEN;
  int t = i - b * T_LEN;
  float sf = s_ws[t * B_SZ + b];
  float sb = s_ws[T_LEN * B_SZ + t * B_SZ + b];
  y[i] = sigmoid_fast(0.5f * (sf + sb) + b_fc[0]);
}

extern "C" void kernel_launch(void* const* d_in, const int* in_sizes, int n_in,
                              void* d_out, int out_size, void* d_ws, size_t ws_size,
                              hipStream_t stream) {
  const float* x      = (const float*)d_in[0];
  const float* w_ih_f = (const float*)d_in[1];
  const float* w_hh_f = (const float*)d_in[2];
  const float* b_ih_f = (const float*)d_in[3];
  const float* b_hh_f = (const float*)d_in[4];
  const float* w_ih_b = (const float*)d_in[5];
  const float* w_hh_b = (const float*)d_in[6];
  const float* b_ih_b = (const float*)d_in[7];
  const float* b_hh_b = (const float*)d_in[8];
  const float* w_fc   = (const float*)d_in[9];
  const float* b_fc   = (const float*)d_in[10];

  float* out  = (float*)d_out;                 // [B*T] y, then [2*B*H] h_n
  float* s_ws = (float*)d_ws;                  // 2*T*B floats of scratch

  birnn_scan_kernel<<<32, 32, 0, stream>>>(x,
                                           w_ih_f, w_hh_f, b_ih_f, b_hh_f,
                                           w_ih_b, w_hh_b, b_ih_b, b_hh_b,
                                           w_fc,
                                           s_ws, out + B_SZ * T_LEN);

  birnn_combine_kernel<<<(B_SZ * T_LEN + 255) / 256, 256, 0, stream>>>(s_ws, b_fc, out);
}
